// Mamba2D_73650099192502
// MI455X (gfx1250) — matrix operand order repaired and checked by reading, not compiled, not run on hardware
//
#include <hip/hip_runtime.h>
#include <hip/hip_bf16.h>

// ---------------------------------------------------------------------------
// Vision-Mamba forward for MI455X (gfx1250, wave32).
// GEMMs: v_wmma_f32_16x16x32_f16.  Block tile 32x64, 4 waves; each wave owns
// a 16x32 output tile = 2 WMMAs sharing one A fragment (raises WMMA:LDS ratio).
// f32 LDS tiles are raw copies of global (async global->LDS b128, ASYNCcnt),
// f16 conversion happens at fragment build (v_cvt_pk_f16_f32).
// Selective scans: register-resident sequential recurrence (N=16 states).
// ---------------------------------------------------------------------------

typedef __attribute__((ext_vector_type(16))) _Float16 v16h;
typedef __attribute__((ext_vector_type(8)))  float    v8f;
typedef __attribute__((ext_vector_type(4)))  int      i32x4;

#define STATE   16
#define DT_RANK 12

#if defined(__has_builtin)
#  if __has_builtin(__builtin_amdgcn_global_load_async_to_lds_b128)
#    define HAVE_ASYNC_COPY 1
#  endif
#endif
#ifndef HAVE_ASYNC_COPY
#  define HAVE_ASYNC_COPY 0
#endif

__device__ __forceinline__ void wait_async0() {
#if defined(__has_builtin) && __has_builtin(__builtin_amdgcn_s_wait_asynccnt)
  __builtin_amdgcn_s_wait_asynccnt(0);
#else
  asm volatile("s_wait_asynccnt 0" ::: "memory");
#endif
}

#if HAVE_ASYNC_COPY
__device__ __forceinline__ void async_copy16(const float* gsrc, float* ldst) {
  __builtin_amdgcn_global_load_async_to_lds_b128(
      (__attribute__((address_space(1))) i32x4*)(gsrc),
      (__attribute__((address_space(3))) i32x4*)(ldst),
      0, 0);
}
#endif

__device__ __forceinline__ float silu_f(float x) { return x / (1.0f + __expf(-x)); }
__device__ __forceinline__ float softplus_f(float x) {
  return (x > 20.0f) ? x : __logf(1.0f + __expf(x));
}

// ---------------------------------------------------------------------------
// Generic tiled GEMM: C[M,N] = act(A[M,K](lda) @ W[K,N] + bias) (+ res)
// Block = 128 threads = 4 waves arranged 2(M) x 2(N); wave tile = 16x32
// (2 WMMAs, shared A fragment).  Block tile 32x64, K step 32.
// ACT: 0=none, 1=softplus.  All call sites have lda%4==0 and N%4==0, so
// every 4-float quad is 16B aligned.
// ---------------------------------------------------------------------------
template <int ACT, bool HAS_BIAS, bool HAS_RES>
__global__ __launch_bounds__(128) void k_gemm(
    const float* __restrict__ A, int lda,
    const float* __restrict__ Wt,          // [K,N] row-major
    const float* __restrict__ bias,        // [N] or nullptr
    const float* __restrict__ res,         // [M,N] or nullptr
    float* __restrict__ C,
    int M, int K, int N)
{
  __shared__ __align__(16) float As[32][32];   // [m][k]  raw f32 copy
  __shared__ __align__(16) float Bs[32][64];   // [k][n]  raw f32 copy

  const int tid   = threadIdx.x;
  const int lane  = tid & 31;
  const int wv    = tid >> 5;
  const int hi    = lane >> 4;     // lane half (ISA 7.12.2 layouts)
  const int l15   = lane & 15;
  const int mBase = blockIdx.y * 32;
  const int nBase = blockIdx.x * 64;
  const int wm    = (wv >> 1) * 16;        // 0 or 16
  const int wn    = (wv & 1) * 32;         // 0 or 32

  v8f acc[2];
  acc[0] = {};
  acc[1] = {};

  for (int k0 = 0; k0 < K; k0 += 32) {
    const bool fullA = (mBase + 32 <= M) && (k0 + 32 <= K);
    const bool fullB = (nBase + 64 <= N) && (k0 + 32 <= K);
    bool used_async = false;

    // ---- A tile: As[r][kk] = A[mBase+r][k0+kk]  (1024 elems, 2 quads/thread)
    if (fullA) {
#if HAVE_ASYNC_COPY
      #pragma unroll
      for (int q = 0; q < 2; ++q) {
        int lin = q * 512 + tid * 4;
        int r = lin >> 5, kk = lin & 31;
        async_copy16(A + (size_t)(mBase + r) * lda + k0 + kk, &As[r][kk]);
      }
      used_async = true;
#else
      #pragma unroll
      for (int q = 0; q < 2; ++q) {
        int lin = q * 512 + tid * 4;
        int r = lin >> 5, kk = lin & 31;
        float4 v = *(const float4*)(A + (size_t)(mBase + r) * lda + k0 + kk);
        *(float4*)(&As[r][kk]) = v;
      }
#endif
    } else {
      // Edge tile: clamp address + select (no exec-mask divergence).
      #pragma unroll
      for (int q = 0; q < 2; ++q) {
        int lin = q * 512 + tid * 4;
        int r = lin >> 5, kk = lin & 31;
        int gm = mBase + r;
        float vv[4];
        #pragma unroll
        for (int j = 0; j < 4; ++j) {
          int gk = k0 + kk + j;
          bool ok = (gm < M) && (gk < K);
          size_t offs = (size_t)(ok ? gm : 0) * lda + (size_t)(ok ? gk : 0);
          float t = A[offs];
          vv[j] = ok ? t : 0.0f;
        }
        *(float4*)(&As[r][kk]) = make_float4(vv[0], vv[1], vv[2], vv[3]);
      }
    }

    // ---- B tile: Bs[r][kk] = Wt[k0+r][nBase+kk]  (2048 elems, 4 quads/thread)
    if (fullB) {
#if HAVE_ASYNC_COPY
      #pragma unroll
      for (int q = 0; q < 4; ++q) {
        int lin = q * 512 + tid * 4;
        int r = lin >> 6, kk = lin & 63;
        async_copy16(Wt + (size_t)(k0 + r) * N + nBase + kk, &Bs[r][kk]);
      }
      used_async = true;
#else
      #pragma unroll
      for (int q = 0; q < 4; ++q) {
        int lin = q * 512 + tid * 4;
        int r = lin >> 6, kk = lin & 63;
        float4 v = *(const float4*)(Wt + (size_t)(k0 + r) * N + nBase + kk);
        *(float4*)(&Bs[r][kk]) = v;
      }
#endif
    } else {
      #pragma unroll
      for (int q = 0; q < 4; ++q) {
        int lin = q * 512 + tid * 4;
        int r = lin >> 6, kk = lin & 63;
        int gk = k0 + r;
        float vv[4];
        #pragma unroll
        for (int j = 0; j < 4; ++j) {
          int gn = nBase + kk + j;
          bool ok = (gk < K) && (gn < N);
          size_t offs = (size_t)(ok ? gk : 0) * N + (size_t)(ok ? gn : 0);
          float t = Wt[offs];
          vv[j] = ok ? t : 0.0f;
        }
        *(float4*)(&Bs[r][kk]) = make_float4(vv[0], vv[1], vv[2], vv[3]);
      }
    }

    if (used_async) wait_async0();
    __syncthreads();

    // ---- fragment build (ISA 7.12.2): one A frag, two B frags, two WMMAs ----
    v16h af;
    #pragma unroll
    for (int j = 0; j < 8; ++j) {
      // A 16x32 f16 layout: lane half selects K offset 8; vgprs 4..7 add +16.
      int ka = ((j & 4) ? 16 : 0) + hi * 8 + 2 * (j & 3);
      af[2 * j]     = (_Float16)As[wm + l15][ka];
      af[2 * j + 1] = (_Float16)As[wm + l15][ka + 1];
    }
    #pragma unroll
    for (int sub = 0; sub < 2; ++sub) {
      v16h bf;
      const int c = wn + sub * 16 + l15;
      #pragma unroll
      for (int j = 0; j < 8; ++j) {
        // B 32x16 f16 layout: lane half selects K offset 8; vgprs 4..7 add +16.
        int kb = ((j & 4) ? 16 : 0) + hi * 8 + 2 * (j & 3);
        bf[2 * j]     = (_Float16)Bs[kb][c];
        bf[2 * j + 1] = (_Float16)Bs[kb + 1][c];
      }
      acc[sub] = __builtin_amdgcn_wmma_f32_16x16x32_f16(
          /*neg_a=*/false, af, /*neg_b=*/false, bf,
          /*c_mod=*/(short)0, acc[sub], /*reuse_a=*/false, /*reuse_b=*/false);
    }
    __syncthreads();
  }

  // C/D layout: VGPR r -> M = r + 8*hi ; N = lane&15.
  #pragma unroll
  for (int sub = 0; sub < 2; ++sub) {
    #pragma unroll
    for (int r = 0; r < 8; ++r) {
      int gm = mBase + wm + hi * 8 + r;
      int gn = nBase + wn + sub * 16 + l15;
      if (gm < M && gn < N) {
        float v = acc[sub][r];
        if (HAS_BIAS) v += bias[gn];
        if (ACT == 1) v = softplus_f(v);
        if (HAS_RES) v += res[(size_t)gm * N + gn];
        C[(size_t)gm * N + gn] = v;
      }
    }
  }
}

// ---------------------------------------------------------------------------
// Patchify p=4 on the input image: out[(b,h0,w0), i*12+j*3+c] = x[b,4h0+i,4w0+j,c]
// ---------------------------------------------------------------------------
__global__ void k_patch4(const float* __restrict__ x, float* __restrict__ out,
                         int B, int H, int W)
{
  int i = blockIdx.x * blockDim.x + threadIdx.x;
  int Ho = H / 4, Wo = W / 4;
  int total = B * Ho * Wo * 48;
  if (i >= total) return;
  int k = i % 48;
  int row = i / 48;
  int w0 = row % Wo; int t = row / Wo;
  int h0 = t % Ho;   int b = t / Ho;
  int c = k % 3; int j = (k / 3) % 4; int pi = k / 12;
  out[i] = x[(((size_t)b * H + (h0 * 4 + pi)) * W + (w0 * 4 + j)) * 3 + c];
}

// Patchify p=2 on feature map [B,H,W,D] -> [B*H/2*W/2, 4D]
__global__ void k_patch2(const float* __restrict__ h, float* __restrict__ out,
                         int B, int H, int W, int D)
{
  int i = blockIdx.x * blockDim.x + threadIdx.x;
  int Ho = H / 2, Wo = W / 2, K4 = 4 * D;
  int total = B * Ho * Wo * K4;
  if (i >= total) return;
  int k = i % K4;
  int row = i / K4;
  int w0 = row % Wo; int t = row / Wo;
  int h0 = t % Ho;   int b = t / Ho;
  int d = k % D; int j = (k / D) & 1; int pi = k / (2 * D);
  out[i] = h[(((size_t)b * H + (2 * h0 + pi)) * W + (2 * w0 + j)) * D + d];
}

// ---------------------------------------------------------------------------
// Row-wise LayerNorm, block(256) per row, width <= 768, safe for src==dst.
// ---------------------------------------------------------------------------
__global__ __launch_bounds__(256) void k_layernorm(
    const float* __restrict__ src, float* __restrict__ dst,
    const float* __restrict__ g, const float* __restrict__ b, int Dw)
{
  __shared__ float sbuf[8];
  __shared__ float s_mean, s_rstd;
  int row = blockIdx.x;
  const float* p = src + (size_t)row * Dw;
  float vals[3];
  int nper = (Dw + 255) / 256;              // <= 3
  float sum = 0.0f;
  for (int t = 0; t < nper; ++t) {
    int idx = threadIdx.x + t * 256;
    float v = (idx < Dw) ? p[idx] : 0.0f;
    vals[t] = v;
    sum += v;
  }
  for (int o = 16; o > 0; o >>= 1) sum += __shfl_down(sum, o, 32);
  if ((threadIdx.x & 31) == 0) sbuf[threadIdx.x >> 5] = sum;
  __syncthreads();
  if (threadIdx.x == 0) {
    float s = 0.0f;
    for (int i = 0; i < 8; ++i) s += sbuf[i];
    s_mean = s / (float)Dw;
  }
  __syncthreads();
  float mean = s_mean;
  float vs = 0.0f;
  for (int t = 0; t < nper; ++t) {
    int idx = threadIdx.x + t * 256;
    if (idx < Dw) { float d = vals[t] - mean; vs += d * d; }
  }
  for (int o = 16; o > 0; o >>= 1) vs += __shfl_down(vs, o, 32);
  if ((threadIdx.x & 31) == 0) sbuf[threadIdx.x >> 5] = vs;
  __syncthreads();
  if (threadIdx.x == 0) {
    float s = 0.0f;
    for (int i = 0; i < 8; ++i) s += sbuf[i];
    s_rstd = rsqrtf(s / (float)Dw + 1e-5f);
  }
  __syncthreads();
  float rstd = s_rstd;
  float* q = dst + (size_t)row * Dw;
  for (int t = 0; t < nper; ++t) {
    int idx = threadIdx.x + t * 256;
    if (idx < Dw) q[idx] = (vals[t] - mean) * rstd * g[idx] + b[idx];
  }
}

// ---------------------------------------------------------------------------
// Depthwise 3x3 SAME conv on xi = xz[...,:E] (row stride 2E), +bias, SiLU.
// ---------------------------------------------------------------------------
__global__ void k_dwconv(const float* __restrict__ xz, const float* __restrict__ cw,
                         const float* __restrict__ cb, float* __restrict__ out,
                         int B, int H, int W, int E)
{
  int i = blockIdx.x * blockDim.x + threadIdx.x;
  int total = B * H * W * E;
  if (i >= total) return;
  int e = i % E; int t = i / E;
  int w = t % W; t /= W;
  int h = t % H; int b = t / H;
  int E2 = 2 * E;
  float acc = cb[e];
  #pragma unroll
  for (int di = 0; di < 3; ++di) {
    int hh = h + di - 1;
    if (hh < 0 || hh >= H) continue;
    #pragma unroll
    for (int dj = 0; dj < 3; ++dj) {
      int ww = w + dj - 1;
      if (ww < 0 || ww >= W) continue;
      acc += xz[(((size_t)b * H + hh) * W + ww) * E2 + e] * cw[(di * 3 + dj) * E + e];
    }
  }
  out[i] = silu_f(acc);
}

// ---------------------------------------------------------------------------
// Selective scan: h_t = exp(dt*A)*h_{t-1} + (dt*u)*B_t ; y_t = <h_t, C_t>.
// One thread per (b,e); 16 states kept in VGPRs. COL selects traversal order.
// proj holds [dt_low(12) | Bm(16) | Cm(16)] per token.
// ---------------------------------------------------------------------------
template <bool COL>
__global__ void k_scan(const float* __restrict__ xi, const float* __restrict__ dt,
                       const float* __restrict__ proj, const float* __restrict__ A_log,
                       float* __restrict__ y, int H, int W, int E)
{
  int e = blockIdx.x * blockDim.x + threadIdx.x;
  int b = blockIdx.y;
  if (e >= E) return;
  const int L = H * W;
  float Ae[STATE], hst[STATE];
  #pragma unroll
  for (int n = 0; n < STATE; ++n) {
    Ae[n]  = -__expf(A_log[e * STATE + n]);
    hst[n] = 0.0f;
  }
  const size_t base = (size_t)b * L;
  for (int t = 0; t < L; ++t) {
    int l;
    if (COL) { int hh = t % H; int w = t / H; l = hh * W + w; }
    else     { l = t; }
    size_t idx = base + l;
    float dtv = dt[idx * E + e];
    float u   = xi[idx * E + e];
    float du  = dtv * u;
    const float* pr = proj + idx * (DT_RANK + 2 * STATE);
    float acc = 0.0f;
    #pragma unroll
    for (int n = 0; n < STATE; ++n) {
      float a  = __expf(dtv * Ae[n]);
      float h2 = a * hst[n] + du * pr[DT_RANK + n];
      hst[n] = h2;
      acc += h2 * pr[DT_RANK + STATE + n];
    }
    y[idx * E + e] = acc;
  }
}

// y_r = (y_r + y_c + xi*D_skip) * silu(z) ; z = xz[..., E:2E]
__global__ void k_combine(float* __restrict__ yr, const float* __restrict__ yc,
                          const float* __restrict__ xi, const float* __restrict__ xz,
                          const float* __restrict__ Dsk, int M, int E)
{
  int i = blockIdx.x * blockDim.x + threadIdx.x;
  if (i >= M * E) return;
  int e = i % E; int m = i / E;
  float z = xz[(size_t)m * (2 * E) + E + e];
  yr[i] = (yr[i] + yc[i] + xi[i] * Dsk[e]) * silu_f(z);
}

__global__ void k_meanpool(const float* __restrict__ h, float* __restrict__ out,
                           int B, int L, int D)
{
  int i = blockIdx.x * blockDim.x + threadIdx.x;
  if (i >= B * D) return;
  int d = i % D; int b = i / D;
  float s = 0.0f;
  for (int l = 0; l < L; ++l) s += h[((size_t)b * L + l) * D + d];
  out[i] = s / (float)L;
}

// Final LayerNorm(192) + 192x2 head; one block per batch element.
__global__ __launch_bounds__(256) void k_head(
    const float* __restrict__ pooled, const float* __restrict__ g,
    const float* __restrict__ bb, const float* __restrict__ hw,
    const float* __restrict__ hb, float* __restrict__ out, int D)
{
  __shared__ float feat[256];
  __shared__ float sbuf[8];
  __shared__ float s_mean, s_rstd;
  int b = blockIdx.x;
  const float* p = pooled + (size_t)b * D;
  float v = (threadIdx.x < D) ? p[threadIdx.x] : 0.0f;
  float sum = v;
  for (int o = 16; o > 0; o >>= 1) sum += __shfl_down(sum, o, 32);
  if ((threadIdx.x & 31) == 0) sbuf[threadIdx.x >> 5] = sum;
  __syncthreads();
  if (threadIdx.x == 0) {
    float s = 0.0f; for (int i = 0; i < 8; ++i) s += sbuf[i];
    s_mean = s / (float)D;
  }
  __syncthreads();
  float mean = s_mean;
  float d2 = (threadIdx.x < D) ? (v - mean) * (v - mean) : 0.0f;
  for (int o = 16; o > 0; o >>= 1) d2 += __shfl_down(d2, o, 32);
  if ((threadIdx.x & 31) == 0) sbuf[threadIdx.x >> 5] = d2;
  __syncthreads();
  if (threadIdx.x == 0) {
    float s = 0.0f; for (int i = 0; i < 8; ++i) s += sbuf[i];
    s_rstd = rsqrtf(s / (float)D + 1e-5f);
  }
  __syncthreads();
  if (threadIdx.x < D)
    feat[threadIdx.x] = (v - mean) * s_rstd * g[threadIdx.x] + bb[threadIdx.x];
  __syncthreads();
  if (threadIdx.x < 2) {
    float acc = hb[threadIdx.x];
    for (int d = 0; d < D; ++d) acc += feat[d] * hw[d * 2 + threadIdx.x];
    out[b * 2 + threadIdx.x] = acc;
  }
}

// ---------------------------------------------------------------------------
// Host orchestration.
// Input order assumption: top-level setup_inputs() insertion order (x, params),
// params flattened jax-tree style (dict keys sorted, list order preserved):
//   0:x 1:head_b 2:head_w 3:norm_b 4:norm_g
//   per stage s (base 5, +14 for stages with downsample, +11 for last):
//     A_log D_skip W_dt W_in W_out W_xproj conv_b conv_w dt_bias ln_b ln_g
//     [down_ln_b down_ln_g down_w]
//   then stem_b stem_ln_b stem_ln_g stem_w
// ---------------------------------------------------------------------------
extern "C" void kernel_launch(void* const* d_in, const int* in_sizes, int n_in,
                              void* d_out, int out_size, void* d_ws, size_t ws_size,
                              hipStream_t stream)
{
  (void)in_sizes; (void)n_in; (void)out_size; (void)ws_size;
  const float* x      = (const float*)d_in[0];
  const float* head_b = (const float*)d_in[1];
  const float* head_w = (const float*)d_in[2];
  const float* norm_b = (const float*)d_in[3];
  const float* norm_g = (const float*)d_in[4];

  struct Blk {
    const float *A_log, *D_skip, *W_dt, *W_in, *W_out, *W_xproj,
                *conv_b, *conv_w, *dt_bias, *ln_b, *ln_g;
  };
  Blk blk[4];
  const float *dn_ln_b[3], *dn_ln_g[3], *dn_w[3];
  int idx = 5;
  for (int s = 0; s < 4; ++s) {
    blk[s].A_log   = (const float*)d_in[idx + 0];
    blk[s].D_skip  = (const float*)d_in[idx + 1];
    blk[s].W_dt    = (const float*)d_in[idx + 2];
    blk[s].W_in    = (const float*)d_in[idx + 3];
    blk[s].W_out   = (const float*)d_in[idx + 4];
    blk[s].W_xproj = (const float*)d_in[idx + 5];
    blk[s].conv_b  = (const float*)d_in[idx + 6];
    blk[s].conv_w  = (const float*)d_in[idx + 7];
    blk[s].dt_bias = (const float*)d_in[idx + 8];
    blk[s].ln_b    = (const float*)d_in[idx + 9];
    blk[s].ln_g    = (const float*)d_in[idx + 10];
    idx += 11;
    if (s < 3) {
      dn_ln_b[s] = (const float*)d_in[idx + 0];
      dn_ln_g[s] = (const float*)d_in[idx + 1];
      dn_w[s]    = (const float*)d_in[idx + 2];
      idx += 3;
    }
  }
  const float* stem_b    = (const float*)d_in[idx + 0];
  const float* stem_ln_b = (const float*)d_in[idx + 1];
  const float* stem_ln_g = (const float*)d_in[idx + 2];
  const float* stem_w    = (const float*)d_in[idx + 3];

  // --- workspace bump allocator (floats) ---
  size_t off = 0;
  auto alloc = [&](size_t nfloats) -> float* {
    float* p = (float*)((char*)d_ws + off);
    off += nfloats * sizeof(float);
    off = (off + 255) & ~(size_t)255;
    return p;
  };
  const int Bn = 4;
  const int Hs[4] = {56, 28, 14, 7};
  const int Ds[4] = {96, 192, 192, 192};
  const int M0 = Bn * 56 * 56;                 // 12544 tokens (stage 0)

  float* h      = alloc(1204224);              // max M*D
  float* hn     = alloc(1204224);
  float* a_stem = alloc((size_t)M0 * 48);
  float* xz     = alloc(4816896);              // max M*2E
  float* xi     = alloc(2408448);              // max M*E
  float* dt     = alloc(2408448);
  float* y_r    = alloc(2408448);
  float* y_c    = alloc(2408448);
  float* proj   = alloc((size_t)M0 * (DT_RANK + 2 * STATE));
  float* m_p    = alloc(1204224);              // max (M/4)*4D
  float* pooled = alloc(Bn * 192);

  // Block tile: 32 (M) x 64 (N)
  auto gdim = [](int M, int N) { return dim3((unsigned)((N + 63) / 64), (unsigned)((M + 31) / 32)); };

  // ---- stem: patchify(4) -> GEMM(48->96)+bias -> LayerNorm ----
  {
    int tot = M0 * 48;
    k_patch4<<<(tot + 255) / 256, 256, 0, stream>>>(x, a_stem, Bn, 224, 224);
    k_gemm<0, true, false><<<gdim(M0, 96), 128, 0, stream>>>(
        a_stem, 48, stem_w, stem_b, nullptr, h, M0, 48, 96);
    k_layernorm<<<M0, 256, 0, stream>>>(h, h, stem_ln_g, stem_ln_b, 96);
  }

  // ---- stages ----
  for (int s = 0; s < 4; ++s) {
    const int H = Hs[s], W = Hs[s], D = Ds[s], E = 2 * D, N2 = 2 * E;
    const int M = Bn * H * W;
    const Blk& p = blk[s];

    // pre-norm
    k_layernorm<<<M, 256, 0, stream>>>(h, hn, p.ln_g, p.ln_b, D);
    // input projection: xz = hn @ W_in  [M, 2E]
    k_gemm<0, false, false><<<gdim(M, N2), 128, 0, stream>>>(
        hn, D, p.W_in, nullptr, nullptr, xz, M, D, N2);
    // depthwise conv + SiLU on xi half
    k_dwconv<<<(M * E + 255) / 256, 256, 0, stream>>>(
        xz, p.conv_w, p.conv_b, xi, Bn, H, W, E);
    // proj = xi @ W_xproj  [M, 44]
    k_gemm<0, false, false><<<gdim(M, DT_RANK + 2 * STATE), 128, 0, stream>>>(
        xi, E, p.W_xproj, nullptr, nullptr, proj, M, E, DT_RANK + 2 * STATE);
    // dt = softplus(proj[:, :12] @ W_dt + dt_bias)  [M, E]
    k_gemm<1, true, false><<<gdim(M, E), 128, 0, stream>>>(
        proj, DT_RANK + 2 * STATE, p.W_dt, p.dt_bias, nullptr, dt, M, DT_RANK, E);
    // selective scans (row-major and column-major)
    dim3 sg((unsigned)((E + 127) / 128), (unsigned)Bn);
    k_scan<false><<<sg, 128, 0, stream>>>(xi, dt, proj, p.A_log, y_r, H, W, E);
    k_scan<true ><<<sg, 128, 0, stream>>>(xi, dt, proj, p.A_log, y_c, H, W, E);
    // combine + gate
    k_combine<<<(M * E + 255) / 256, 256, 0, stream>>>(
        y_r, y_c, xi, xz, p.D_skip, M, E);
    // output projection with residual: h = h + y @ W_out
    k_gemm<0, false, true><<<gdim(M, D), 128, 0, stream>>>(
        y_r, E, p.W_out, nullptr, h, h, M, E, D);

    // downsample
    if (s < 3) {
      int Ho = H / 2, K4 = 4 * D, Dn = Ds[s + 1];
      int M2 = Bn * Ho * Ho;
      k_patch2<<<(M2 * K4 + 255) / 256, 256, 0, stream>>>(h, m_p, Bn, H, W, D);
      k_layernorm<<<M2, 256, 0, stream>>>(m_p, m_p, dn_ln_g[s], dn_ln_b[s], K4);
      k_gemm<0, false, false><<<gdim(M2, Dn), 128, 0, stream>>>(
          m_p, K4, dn_w[s], nullptr, nullptr, h, M2, K4, Dn);
    }
  }

  // ---- pool + final norm + head ----
  k_meanpool<<<(Bn * 192 + 255) / 256, 256, 0, stream>>>(h, pooled, Bn, 7 * 7, 192);
  k_head<<<Bn, 256, 0, stream>>>(pooled, norm_g, norm_b, head_w, head_b,
                                 (float*)d_out, 192);
}
